// sLSTMblock_54872502174275
// MI455X (gfx1250) — compile-verified, matrix operand order checked by reading, not run on hardware
//
#include <hip/hip_runtime.h>
#include <hip/hip_bf16.h>
#include <math.h>

// ---------------------------------------------------------------------------
// sLSTM block for MI455X (gfx1250, wave32, WMMA).
// All matrix math runs on v_wmma_f32_16x16x32_bf16 (bf16 in, f32 accumulate).
// Main GEMM uses CDNA5 async global->LDS staging (ASYNCcnt) + double buffer.
// Sequential scan runs as a persistent 32-WG kernel with a software grid
// barrier; R matrices (8 MB bf16) stay resident in the 192 MB L2.
// ---------------------------------------------------------------------------

typedef __bf16 bf16_t;
typedef __attribute__((ext_vector_type(16))) __bf16 v16bf;
typedef __attribute__((ext_vector_type(8)))  float  v8f;
typedef __attribute__((address_space(3))) char lds_char;   // 32-bit LDS ptr

#define B_    8
#define T_    2048
#define D_    1024
#define F_    1365
#define FP_   1408      // F padded to 22*64 tiles; padded weight rows are zero
#define BT_   (B_*T_)
#define KCONV 4

#define MT 128
#define NT 64
#define KT 32
#define SCAN_WGS 32

// ---- WMMA fragment loader -------------------------------------------------
// CDNA5 16-bit A/B layout (ISA 7.12.2): lanes 0-15 -> M(or N)=lane, K chunks
// {0..7, 16..23}; lanes 16-31 -> same row index, K chunks {8..15, 24..31}.
static __device__ inline v16bf load_frag16(const bf16_t* rowp, int hi) {
  const uint4* p = (const uint4*)(rowp + hi * 8);   // 16B aligned by layout
  union { uint4 u[2]; v16bf v; } cvt;
  cvt.u[0] = p[0];      // K chunk 0 (8 bf16)
  cvt.u[1] = p[2];      // K chunk +16 elements
  return cvt.v;
}

static __device__ inline v8f wmma_bf16(v16bf a, v16bf b, v8f c) {
  return __builtin_amdgcn_wmma_f32_16x16x32_bf16(
      false, a, false, b, (short)0, c, false, false);
}

// ---- CDNA5 async global->LDS copy (ASYNCcnt-tracked, no VGPR staging) -----
// The LDS destination is passed as an address-space(3) pointer: its 32-bit
// value is the wave LDS offset (what VDST wants), and passing the pointer
// makes the __shared__ object escape so the compiler keeps the ds_load reads.
static __device__ inline void async_ld128(lds_char* lp, const void* gaddr) {
  asm volatile("global_load_async_to_lds_b128 %0, %1, off"
               :: "v"(lp), "v"((unsigned long long)gaddr)
               : "memory");
}
static __device__ inline void wait_async_le3() {
  asm volatile("s_wait_asynccnt 0x3" ::: "memory");
}
static __device__ inline void wait_async_0() {
  asm volatile("s_wait_asynccnt 0x0" ::: "memory");
}

// ---- generic fused GEMM: C = epi(A[MxK] * W[NxK]^T) -----------------------
// Double-buffered LDS tiles filled by async copies; 4x2 wave grid, each wave
// computes a 32x32 block via 4 chained v_wmma per K slab.
// act: 0=none, 1=exact GELU. gate: optional bf16 multiplier (GEGLU).
// resid: optional f32 residual add. outF/outB optional outputs.
__global__ __launch_bounds__(256)
void gemm_bf16_kernel(const bf16_t* __restrict__ A, int lda,
                      const bf16_t* __restrict__ W, int ldw,
                      const float* __restrict__ bias, int biasN,
                      const bf16_t* __restrict__ gate,
                      const float* __restrict__ resid,
                      float* __restrict__ outF, bf16_t* __restrict__ outB,
                      int ldc, int Kdim, int act)
{
  __shared__ __align__(16) bf16_t lds[2 * (MT + NT) * KT];   // 2 x 12 KB
  const int tid  = threadIdx.x;
  const int lane = tid & 31, wid = tid >> 5;
  const int l15  = lane & 15, hi = lane >> 4;
  const int m0 = blockIdx.x * MT, n0 = blockIdx.y * NT;
  const int wm = (wid >> 1) * 32, wn = (wid & 1) * 32;   // 4x2 wave grid

  // per-thread staging sources: 3 x b128 per K slab (2 A rows + 1 B row)
  const int srow = tid >> 2, sq = tid & 3;
  const bf16_t* apSrc0 = A + (size_t)(m0 + srow) * lda + sq * 8;
  const bf16_t* apSrc1 = apSrc0 + (size_t)64 * lda;
  const bf16_t* bpSrc  = W + (size_t)(n0 + srow) * ldw + sq * 8;
  lds_char* ldsb = (lds_char*)lds;                // AS(3) base, offset 0
  const unsigned BUFB  = (MT + NT) * KT * 2;      // 12288 B per buffer
  const unsigned aOff0 = (unsigned)tid * 16u;
  const unsigned aOff1 = (unsigned)(tid + 256) * 16u;
  const unsigned bOff  = (unsigned)(MT * KT * 2) + (unsigned)tid * 16u;

  v8f acc00 = {}, acc01 = {}, acc10 = {}, acc11 = {};

  const int niter = Kdim / KT;
  // prologue: tile 0 -> buffer 0
  async_ld128(ldsb + aOff0, apSrc0);
  async_ld128(ldsb + aOff1, apSrc1);
  async_ld128(ldsb + bOff,  bpSrc);

  for (int it = 0; it < niter; ++it) {
    if (it + 1 < niter) {
      // issue next tile into the other buffer, then wait only for the
      // current tile's 3 ops (async loads complete in order)
      const unsigned bo = ((it + 1) & 1) * BUFB;
      const size_t kb = (size_t)(it + 1) * KT * sizeof(bf16_t);
      async_ld128(ldsb + bo + aOff0, (const char*)apSrc0 + kb);
      async_ld128(ldsb + bo + aOff1, (const char*)apSrc1 + kb);
      async_ld128(ldsb + bo + bOff,  (const char*)bpSrc  + kb);
      wait_async_le3();
    } else {
      wait_async_0();
    }
    __syncthreads();

    const bf16_t* As = lds + (it & 1) * (MT + NT) * KT;
    const bf16_t* Bs = As + MT * KT;
    v16bf a0 = load_frag16(As + (wm + l15) * KT, hi);
    v16bf a1 = load_frag16(As + (wm + 16 + l15) * KT, hi);
    v16bf b0 = load_frag16(Bs + (wn + l15) * KT, hi);
    v16bf b1 = load_frag16(Bs + (wn + 16 + l15) * KT, hi);
    acc00 = wmma_bf16(a0, b0, acc00);
    acc01 = wmma_bf16(a0, b1, acc01);
    acc10 = wmma_bf16(a1, b0, acc10);
    acc11 = wmma_bf16(a1, b1, acc11);
    __syncthreads();   // readers done before this buffer is refilled
  }

  // epilogue: C layout: lanes 0-15: N=lane, VGPR r -> M=r; lanes 16-31:
  // N=lane-16, VGPR r -> M=r+8.
  v8f* accs[2][2] = {{&acc00, &acc01}, {&acc10, &acc11}};
#pragma unroll
  for (int mi = 0; mi < 2; ++mi)
#pragma unroll
    for (int ni = 0; ni < 2; ++ni) {
      int col  = n0 + wn + ni * 16 + l15;
      int rowb = m0 + wm + mi * 16 + hi * 8;
#pragma unroll
      for (int r = 0; r < 8; ++r) {
        int row = rowb + r;
        float v = (*accs[mi][ni])[r];
        if (bias) v += (col < biasN) ? bias[col] : 0.f;
        if (act == 1) v = 0.5f * v * (1.f + erff(v * 0.70710678f));
        size_t o = (size_t)row * ldc + col;
        if (gate)  v *= (float)gate[o];
        if (resid) v += resid[o];
        if (outF) outF[o] = v;
        if (outB) outB[o] = (bf16_t)v;
      }
    }
}

// ---- causal conv (K=4) as 4 accumulated shifted GEMMs + swish -------------
// Synchronous staging (rows with t<lag must be zero-filled, which the async
// path cannot produce).
__global__ __launch_bounds__(256)
void conv_swish_kernel(const bf16_t* __restrict__ xnorm,
                       const bf16_t* __restrict__ Wc,  // [lag][Dout][Din]
                       const float* __restrict__ cb,
                       bf16_t* __restrict__ xif)
{
  __shared__ __align__(16) bf16_t As[MT][KT];
  __shared__ __align__(16) bf16_t Bs[NT][KT];
  const int tid  = threadIdx.x;
  const int lane = tid & 31, wid = tid >> 5;
  const int l15  = lane & 15, hi = lane >> 4;
  const int m0 = blockIdx.x * MT, n0 = blockIdx.y * NT;
  const int wm = (wid >> 1) * 32, wn = (wid & 1) * 32;

  v8f acc00 = {}, acc01 = {}, acc10 = {}, acc11 = {};
  uint4* AsV = (uint4*)(&As[0][0]);
  uint4* BsV = (uint4*)(&Bs[0][0]);
  const uint4 zero4 = make_uint4(0u, 0u, 0u, 0u);
  const int srow = tid >> 2, sq = tid & 3;

  for (int lag = 0; lag < KCONV; ++lag) {
    const bf16_t* Wl = Wc + (size_t)lag * D_ * D_;
    const bf16_t* bp = Wl + (size_t)(n0 + srow) * D_ + sq * 8;
    for (int k0 = 0; k0 < D_; k0 += KT) {
      for (int c = tid; c < 512; c += 256) {
        int r = c >> 2, q = c & 3;
        int g = m0 + r;                          // token index; tiles never
        bool valid = (g & (T_ - 1)) >= lag;      // straddle batch (T%MT==0)
        AsV[c] = valid
            ? *(const uint4*)(xnorm + (size_t)(g - lag) * D_ + k0 + q * 8)
            : zero4;
      }
      BsV[tid] = *(const uint4*)(bp + k0);
      __syncthreads();
      v16bf a0 = load_frag16(&As[wm + l15][0], hi);
      v16bf a1 = load_frag16(&As[wm + 16 + l15][0], hi);
      v16bf b0 = load_frag16(&Bs[wn + l15][0], hi);
      v16bf b1 = load_frag16(&Bs[wn + 16 + l15][0], hi);
      acc00 = wmma_bf16(a0, b0, acc00);
      acc01 = wmma_bf16(a0, b1, acc01);
      acc10 = wmma_bf16(a1, b0, acc10);
      acc11 = wmma_bf16(a1, b1, acc11);
      __syncthreads();
    }
  }

  v8f* accs[2][2] = {{&acc00, &acc01}, {&acc10, &acc11}};
#pragma unroll
  for (int mi = 0; mi < 2; ++mi)
#pragma unroll
    for (int ni = 0; ni < 2; ++ni) {
      int col  = n0 + wn + ni * 16 + l15;
      int rowb = m0 + wm + mi * 16 + hi * 8;
#pragma unroll
      for (int r = 0; r < 8; ++r) {
        int row = rowb + r;
        float v = (*accs[mi][ni])[r] + cb[col];
        v = v / (1.f + expf(-v));                      // swish
        xif[(size_t)row * D_ + col] = (bf16_t)v;
      }
    }
}

// ---- layernorm(x) -> bf16 x_norm, plus bf16 cast of raw x -----------------
__global__ __launch_bounds__(256)
void ln_cast_kernel(const float* __restrict__ x, const float* __restrict__ g,
                    const float* __restrict__ b, bf16_t* __restrict__ xnorm,
                    bf16_t* __restrict__ xb)
{
  __shared__ float red[256];
  const int row = blockIdx.x, tid = threadIdx.x;
  const float* xr = x + (size_t)row * D_;
  float s = 0.f;
  for (int i = tid; i < D_; i += 256) s += xr[i];
  red[tid] = s; __syncthreads();
  for (int st = 128; st > 0; st >>= 1) {
    if (tid < st) red[tid] += red[tid + st];
    __syncthreads();
  }
  float mu = red[0] * (1.f / D_); __syncthreads();
  float v = 0.f;
  for (int i = tid; i < D_; i += 256) { float d = xr[i] - mu; v += d * d; }
  red[tid] = v; __syncthreads();
  for (int st = 128; st > 0; st >>= 1) {
    if (tid < st) red[tid] += red[tid + st];
    __syncthreads();
  }
  float rs = rsqrtf(red[0] * (1.f / D_) + 1e-5f);
  for (int i = tid; i < D_; i += 256) {
    float xn = (xr[i] - mu) * rs * g[i] + b[i];
    xnorm[(size_t)row * D_ + i] = (bf16_t)xn;
    if (xb) xb[(size_t)row * D_ + i] = (bf16_t)xr[i];
  }
}

// ---- layernorm(h_seq)*gn + x -> h_skip (f32) ------------------------------
__global__ __launch_bounds__(256)
void gn_skip_kernel(const float* __restrict__ hseq, const float* __restrict__ x,
                    const float* __restrict__ g, const float* __restrict__ b,
                    float* __restrict__ hskip)
{
  __shared__ float red[256];
  const int row = blockIdx.x, tid = threadIdx.x;
  const float* hr = hseq + (size_t)row * D_;
  float s = 0.f;
  for (int i = tid; i < D_; i += 256) s += hr[i];
  red[tid] = s; __syncthreads();
  for (int st = 128; st > 0; st >>= 1) {
    if (tid < st) red[tid] += red[tid + st];
    __syncthreads();
  }
  float mu = red[0] * (1.f / D_); __syncthreads();
  float v = 0.f;
  for (int i = tid; i < D_; i += 256) { float d = hr[i] - mu; v += d * d; }
  red[tid] = v; __syncthreads();
  for (int st = 128; st > 0; st >>= 1) {
    if (tid < st) red[tid] += red[tid + st];
    __syncthreads();
  }
  float rs = rsqrtf(red[0] * (1.f / D_) + 1e-5f);
  for (int i = tid; i < D_; i += 256) {
    size_t o = (size_t)row * D_ + i;
    hskip[o] = (hr[i] - mu) * rs * g[i] + b[i] + x[o];
  }
}

// ---- software grid barrier (persistent scan kernel) -----------------------
static __device__ inline void grid_sync(unsigned* cnt, volatile unsigned* gen,
                                        unsigned nwg) {
  __syncthreads();
  if (threadIdx.x == 0) {
    unsigned g = *gen;
    __threadfence();
    if (atomicAdd(cnt, 1u) + 1u == nwg) {
      atomicExch(cnt, 0u);
      __threadfence();
      atomicAdd((unsigned*)gen, 1u);
    } else {
      while (*gen == g) __builtin_amdgcn_s_sleep(1);
    }
  }
  __syncthreads();
}

// ---- sequential sLSTM scan: 32 WGs x 256 threads, persistent --------------
// 256 waves: wave w -> gate w>>6, column frag (w&63)*16. h padded to 16 rows
// (rows 8..15 zero) for the fixed 16x16 WMMA shape; per step each wave runs a
// 32-deep chain of v_wmma over K=1024. R matrices stream from L2.
__global__ __launch_bounds__(256)
void scan_kernel(const float* __restrict__ zx, const float* __restrict__ ix,
                 const float* __restrict__ fx, const float* __restrict__ ox,
                 const bf16_t* __restrict__ Rz, const bf16_t* __restrict__ Ri,
                 const bf16_t* __restrict__ Rf, const bf16_t* __restrict__ Ro,
                 float* __restrict__ recur, float* __restrict__ hS,
                 float* __restrict__ cS, float* __restrict__ nS,
                 float* __restrict__ mS, bf16_t* __restrict__ hB,
                 float* __restrict__ h_seq, unsigned* __restrict__ bar)
{
  const int tid  = threadIdx.x;
  const int gtid = blockIdx.x * 256 + tid;            // 0..8191 == B_*D_
  volatile unsigned* gen = bar + 1;

  // zero recurrent state (bar[0..1] zeroed by a prior kernel)
  hS[gtid] = 0.f; cS[gtid] = 0.f; nS[gtid] = 0.f; mS[gtid] = 0.f;
  hB[gtid] = (bf16_t)0.f; hB[8192 + gtid] = (bf16_t)0.f;
  grid_sync(bar, gen, SCAN_WGS);

  const int wave  = blockIdx.x * 8 + (tid >> 5);      // 0..255
  const int gidx  = wave >> 6;                        // 0=z 1=i 2=f 3=o
  const int ncol0 = (wave & 63) * 16;
  const bf16_t* R = (gidx == 0) ? Rz : (gidx == 1) ? Ri : (gidx == 2) ? Rf : Ro;
  const int lane = tid & 31, l15 = lane & 15, hi = lane >> 4;
  const int bb = gtid >> 10, dd = gtid & 1023;

  const bf16_t* hrow = hB + (size_t)l15 * D_;
  const bf16_t* wrow = R + (size_t)(ncol0 + l15) * D_;

  for (int t = 0; t < T_; ++t) {
    v8f acc = {};
#pragma unroll 4
    for (int k0 = 0; k0 < D_; k0 += KT)
      acc = wmma_bf16(load_frag16(hrow + k0, hi), load_frag16(wrow + k0, hi), acc);
    if (hi == 0) {                                     // rows 0..7 are real
      int col = ncol0 + l15;
#pragma unroll
      for (int r = 0; r < 8; ++r)
        recur[gidx * (B_ * D_) + r * D_ + col] = acc[r];
    }
    grid_sync(bar, gen, SCAN_WGS);

    {   // pointwise sLSTM cell: one thread per (b,d)
      size_t o = ((size_t)bb * T_ + t) * D_ + dd;
      float zb = zx[o] + recur[0 * (B_ * D_) + gtid];
      float ib = ix[o] + recur[1 * (B_ * D_) + gtid];
      float fb = fx[o] + recur[2 * (B_ * D_) + gtid];
      float ob = ox[o] + recur[3 * (B_ * D_) + gtid];
      float mo = mS[gtid];
      float mn = fmaxf(fb + mo, ib);
      float iv = expf(ib - mn);
      float fv = expf(fb + mo - mn);
      float cn = fv * cS[gtid] + iv * tanhf(zb);
      float nn = fv * nS[gtid] + iv;
      float hn = (1.f / (1.f + expf(-ob))) * (cn / (nn + 1e-8f));
      cS[gtid] = cn; nS[gtid] = nn; mS[gtid] = mn; hS[gtid] = hn;
      h_seq[o] = hn;
      hB[gtid] = (bf16_t)hn;
    }
    grid_sync(bar, gen, SCAN_WGS);
  }
}

// ---- small utility kernels ------------------------------------------------
__global__ void cast_bf16_kernel(const float* __restrict__ s,
                                 bf16_t* __restrict__ d, int n) {
  int i = blockIdx.x * 256 + threadIdx.x;
  if (i < n) d[i] = (bf16_t)s[i];
}
// conv_w (Dout,Din,K) -> [lag][Dout][Din] with lag = (K-1)-kk (causal shift)
__global__ void conv_rearrange_kernel(const float* __restrict__ cw,
                                      bf16_t* __restrict__ dst) {
  int i = blockIdx.x * 256 + threadIdx.x;
  if (i < KCONV * D_ * D_) {
    int lag = i / (D_ * D_);
    int rem = i - lag * (D_ * D_);
    int o = rem / D_, in = rem - o * D_;
    dst[i] = (bf16_t)cw[((size_t)o * D_ + in) * KCONV + (KCONV - 1 - lag)];
  }
}
__global__ void pad_rows_kernel(const float* __restrict__ s,
                                bf16_t* __restrict__ d, int rs, int rd, int c) {
  int i = blockIdx.x * 256 + threadIdx.x;
  if (i < rd * c) {
    int r = i / c;
    d[i] = (r < rs) ? (bf16_t)s[(size_t)r * c + (i - r * c)] : (bf16_t)0.f;
  }
}
__global__ void pad_cols_kernel(const float* __restrict__ s,
                                bf16_t* __restrict__ d, int rows, int cs, int cd) {
  int i = blockIdx.x * 256 + threadIdx.x;
  if (i < rows * cd) {
    int r = i / cd, c = i - r * cd;
    d[i] = (c < cs) ? (bf16_t)s[(size_t)r * cs + c] : (bf16_t)0.f;
  }
}
__global__ void zero_bar_kernel(unsigned* bar) {
  if (threadIdx.x < 2) bar[threadIdx.x] = 0u;
}
__global__ void tail_kernel(const float* __restrict__ hS, const float* __restrict__ cS,
                            const float* __restrict__ nS, const float* __restrict__ mS,
                            float* __restrict__ out) {
  int i = blockIdx.x * 256 + threadIdx.x;
  if (i < B_ * D_) {
    out[i] = hS[i];
    out[B_ * D_ + i] = cS[i];
    out[2 * B_ * D_ + i] = nS[i];
    out[3 * B_ * D_ + i] = mS[i];
  }
}

// ---------------------------------------------------------------------------
extern "C" void kernel_launch(void* const* d_in, const int* in_sizes, int n_in,
                              void* d_out, int out_size, void* d_ws, size_t ws_size,
                              hipStream_t stream)
{
  (void)in_sizes; (void)n_in; (void)out_size; (void)ws_size;
  const float* x    = (const float*)d_in[0];
  const float* Wz   = (const float*)d_in[1];  const float* bz = (const float*)d_in[2];
  const float* Wi   = (const float*)d_in[3];  const float* bi = (const float*)d_in[4];
  const float* Wf   = (const float*)d_in[5];  const float* bf = (const float*)d_in[6];
  const float* Wo   = (const float*)d_in[7];  const float* bo = (const float*)d_in[8];
  const float* Rz   = (const float*)d_in[9];  const float* Ri = (const float*)d_in[10];
  const float* Rf   = (const float*)d_in[11]; const float* Ro = (const float*)d_in[12];
  const float* cw   = (const float*)d_in[13]; const float* cb = (const float*)d_in[14];
  const float* ln_g = (const float*)d_in[15]; const float* ln_b = (const float*)d_in[16];
  const float* gn_g = (const float*)d_in[17]; const float* gn_b = (const float*)d_in[18];
  const float* lw   = (const float*)d_in[19]; const float* lb = (const float*)d_in[20];
  const float* rw   = (const float*)d_in[21]; const float* rb = (const float*)d_in[22];
  const float* ow   = (const float*)d_in[23]; const float* ob = (const float*)d_in[24];
  float* out = (float*)d_out;

  char* ws = (char*)d_ws;
  size_t off = 0;
  auto alloc = [&](size_t bytes) -> char* {
    off = (off + 255) & ~(size_t)255;
    char* p = ws + off; off += bytes; return p;
  };

  bf16_t* xB   = (bf16_t*)alloc((size_t)BT_ * D_ * 2);
  bf16_t* xnB  = (bf16_t*)alloc((size_t)BT_ * D_ * 2);
  bf16_t* xifB = (bf16_t*)alloc((size_t)BT_ * D_ * 2);
  bf16_t* zlB  = (bf16_t*)alloc((size_t)BT_ * FP_ * 2);
  bf16_t* zB   = (bf16_t*)alloc((size_t)BT_ * FP_ * 2);
  bf16_t* WzB  = (bf16_t*)alloc((size_t)D_ * D_ * 2);
  bf16_t* WiB  = (bf16_t*)alloc((size_t)D_ * D_ * 2);
  bf16_t* WfB  = (bf16_t*)alloc((size_t)D_ * D_ * 2);
  bf16_t* WoB  = (bf16_t*)alloc((size_t)D_ * D_ * 2);
  bf16_t* RzB  = (bf16_t*)alloc((size_t)D_ * D_ * 2);
  bf16_t* RiB  = (bf16_t*)alloc((size_t)D_ * D_ * 2);
  bf16_t* RfB  = (bf16_t*)alloc((size_t)D_ * D_ * 2);
  bf16_t* RoB  = (bf16_t*)alloc((size_t)D_ * D_ * 2);
  bf16_t* WcB  = (bf16_t*)alloc((size_t)KCONV * D_ * D_ * 2);
  bf16_t* lwB  = (bf16_t*)alloc((size_t)FP_ * D_ * 2);
  bf16_t* rwB  = (bf16_t*)alloc((size_t)FP_ * D_ * 2);
  bf16_t* owB  = (bf16_t*)alloc((size_t)D_ * FP_ * 2);
  float* zxF   = (float*)alloc((size_t)BT_ * D_ * 4);   // reused as h_skip
  float* ixF   = (float*)alloc((size_t)BT_ * D_ * 4);
  float* fxF   = (float*)alloc((size_t)BT_ * D_ * 4);
  float* oxF   = (float*)alloc((size_t)BT_ * D_ * 4);
  float* hseqF = (float*)alloc((size_t)BT_ * D_ * 4);
  float* recur = (float*)alloc((size_t)4 * B_ * D_ * 4);
  float* hS    = (float*)alloc((size_t)B_ * D_ * 4);
  float* cS    = (float*)alloc((size_t)B_ * D_ * 4);
  float* nS    = (float*)alloc((size_t)B_ * D_ * 4);
  float* mS    = (float*)alloc((size_t)B_ * D_ * 4);
  bf16_t* hB   = (bf16_t*)alloc((size_t)16 * D_ * 2);   // 8 real + 8 zero rows
  unsigned* bar = (unsigned*)alloc(256);

  const int nDD = D_ * D_;
  auto cast = [&](const float* s, bf16_t* d, int n) {
    cast_bf16_kernel<<<(n + 255) / 256, 256, 0, stream>>>(s, d, n);
  };

  // --- weight conversion (bf16 for WMMA) ---
  cast(Wz, WzB, nDD); cast(Wi, WiB, nDD); cast(Wf, WfB, nDD); cast(Wo, WoB, nDD);
  cast(Rz, RzB, nDD); cast(Ri, RiB, nDD); cast(Rf, RfB, nDD); cast(Ro, RoB, nDD);
  conv_rearrange_kernel<<<(KCONV * nDD + 255) / 256, 256, 0, stream>>>(cw, WcB);
  pad_rows_kernel<<<(FP_ * D_ + 255) / 256, 256, 0, stream>>>(lw, lwB, F_, FP_, D_);
  pad_rows_kernel<<<(FP_ * D_ + 255) / 256, 256, 0, stream>>>(rw, rwB, F_, FP_, D_);
  pad_cols_kernel<<<(D_ * FP_ + 255) / 256, 256, 0, stream>>>(ow, owB, D_, F_, FP_);

  // --- front-end: LN, conv+swish, gate GEMMs ---
  ln_cast_kernel<<<BT_, 256, 0, stream>>>(x, ln_g, ln_b, xnB, xB);
  conv_swish_kernel<<<dim3(BT_ / MT, D_ / NT), 256, 0, stream>>>(xnB, WcB, cb, xifB);

  auto gemm = [&](const bf16_t* A, int lda, const bf16_t* W, int ldw,
                  const float* bias, int biasN, const bf16_t* gate,
                  const float* resid, float* outF, bf16_t* outB,
                  int ldc, int Kd, int act, int N) {
    gemm_bf16_kernel<<<dim3(BT_ / MT, N / NT), 256, 0, stream>>>(
        A, lda, W, ldw, bias, biasN, gate, resid, outF, outB, ldc, Kd, act);
  };
  gemm(xB,   D_, WzB, D_, bz, D_, nullptr, nullptr, zxF, nullptr, D_, D_, 0, D_);
  gemm(xifB, D_, WiB, D_, bi, D_, nullptr, nullptr, ixF, nullptr, D_, D_, 0, D_);
  gemm(xifB, D_, WfB, D_, bf, D_, nullptr, nullptr, fxF, nullptr, D_, D_, 0, D_);
  gemm(xB,   D_, WoB, D_, bo, D_, nullptr, nullptr, oxF, nullptr, D_, D_, 0, D_);

  // --- sequential scan (persistent) ---
  zero_bar_kernel<<<1, 32, 0, stream>>>(bar);
  scan_kernel<<<SCAN_WGS, 256, 0, stream>>>(zxF, ixF, fxF, oxF,
                                            RzB, RiB, RfB, RoB,
                                            recur, hS, cS, nS, mS, hB, hseqF, bar);

  // --- back-end: gn+skip, GEGLU MLP, output projection + residual ---
  gn_skip_kernel<<<BT_, 256, 0, stream>>>(hseqF, x, gn_g, gn_b, zxF /*h_skip*/);
  gemm(xnB, D_, lwB, D_, lb, F_, nullptr, nullptr, nullptr, zlB, FP_, D_, 0, FP_);
  gemm(xnB, D_, rwB, D_, rb, F_, zlB, nullptr, nullptr, zB,  FP_, D_, 1, FP_);
  gemm(zB, FP_, owB, FP_, ob, D_, nullptr, zxF, out, nullptr, D_, FP_, 0, D_);

  tail_kernel<<<32, 256, 0, stream>>>(hS, cS, nS, mS, out + (size_t)BT_ * D_);
}
